// CsbktModel_18580028522982
// MI455X (gfx1250) — matrix operand order repaired and checked by reading, not compiled, not run on hardware
//
#include <hip/hip_runtime.h>
#include <hip/hip_bf16.h>

typedef __attribute__((ext_vector_type(16))) _Float16 v16h;
typedef __attribute__((ext_vector_type(8)))  float    v8f;

#define NSK 8
#define NS  256
#define NP  1000
#define NA  5
#define NB  64
#define NT  200
#define NROWS (NA * NB)        // 320
#define TSTR 264               // LDS row stride (halves) for trans matrix
#define VSTR 264               // LDS row stride (halves) for V panel (16B aligned)
#define ASTR 260               // LDS row stride (floats) for alpha (16B aligned)

__device__ __forceinline__ float logsigf(float x) {
    // log(sigmoid(x)) = min(x,0) - log1p(exp(-|x|))
    return fminf(x, 0.0f) - log1pf(expf(-fabsf(x)));
}

// ---------------------------------------------------------------------------
// P1: problem_logits[p][s] = sum_k mem[p,k] * (off[k] + slope[k]*bit(s,k))
// ---------------------------------------------------------------------------
__global__ void k_problem_logits(const float* __restrict__ ml,
                                 const float* __restrict__ off,
                                 const float* __restrict__ slope,
                                 const int* __restrict__ test,
                                 float* __restrict__ PL) {
    int p = blockIdx.x;
    int s = threadIdx.x;
    int tst = *test;
    float acc = 0.0f;
#pragma unroll
    for (int k = 0; k < NSK; ++k) {
        float m = 1.0f / (1.0f + expf(-ml[p * NSK + k]));
        if (tst) m = (m > 0.5f) ? 1.0f : 0.0f;
        float bit = (float)((s >> (7 - k)) & 1);
        acc += m * (off[k] + slope[k] * bit);
    }
    PL[p * NS + s] = acc;
}

// ---------------------------------------------------------------------------
// P2: Texp[i][j] = exp( sum_k lpm[k, type(i,j,k)] )  (f16, row i = new state)
// ---------------------------------------------------------------------------
__global__ void k_trans(const float* __restrict__ pL,
                        const float* __restrict__ pF,
                        _Float16* __restrict__ Texp) {
    int i = blockIdx.x;   // new state
    int j = threadIdx.x;  // old state
    float acc = 0.0f;
#pragma unroll
    for (int k = 0; k < NSK; ++k) {
        int di = (i >> (7 - k)) & 1;
        int dj = (j >> (7 - k)) & 1;
        float v;
        if (di == 0) v = (dj == 0) ? logsigf(-pL[k]) : logsigf(pL[k]);
        else         v = (dj == 0) ? logsigf(pF[k])  : logsigf(-pF[k]);
        acc += v;
    }
    Texp[i * NS + j] = (_Float16)expf(acc);
}

// ---------------------------------------------------------------------------
// P3: init[s] = sum_k log_sigmoid( bit(s,k) ? pi[k] : -pi[k] )
// ---------------------------------------------------------------------------
__global__ void k_init(const float* __restrict__ pi, float* __restrict__ initv) {
    int s = threadIdx.x;
    float acc = 0.0f;
#pragma unroll
    for (int k = 0; k < NSK; ++k) {
        float x = ((s >> (7 - k)) & 1) ? pi[k] : -pi[k];
        acc += logsigf(x);
    }
    initv[s] = acc;
}

// ---------------------------------------------------------------------------
// Main recurrence: 20 blocks x 256 threads; block handles 16 rows (same
// ability level), iterates all 200 time steps with T-matrix resident in LDS.
// Per step: phase A (VALU obs/LSE, builds f16 V panel row-major), phase B
// (WMMA: alpha'(256x16) = T(256x256,f16) @ V(256x16,f16), f32 accumulate).
// All LDS layouts chosen so fragment loads/stores are 16B-contiguous.
// ---------------------------------------------------------------------------
__global__ void __launch_bounds__(256) k_forward(const int* __restrict__ corr,
                                                 const int* __restrict__ pseq,
                                                 const float* __restrict__ abil,
                                                 const float* __restrict__ PL,
                                                 const _Float16* __restrict__ Texp,
                                                 const float* __restrict__ initv,
                                                 float* __restrict__ lpws) {
    extern __shared__ char smem[];
    _Float16* Th  = (_Float16*)smem;                                   // [256][TSTR]
    _Float16* Vh  = (_Float16*)(smem + NS * TSTR * 2);                 // [16][VSTR]
    float*    Aal = (float*)(smem + NS * TSTR * 2 + 16 * VSTR * 2);    // [16][ASTR]
    float*    rowm = (float*)(smem + NS * TSTR * 2 + 16 * VSTR * 2 + 16 * ASTR * 4);

    const int tid = threadIdx.x;

    // Load exp(trans) into LDS, 16B per thread-iteration (rows padded to TSTR)
    {
        const uint4* src = (const uint4*)Texp;     // 8 halves per uint4
        for (int idx = tid; idx < (NS * NS) / 8; idx += 256) {
            int i = idx >> 5;            // row (256/8 = 32 chunks per row)
            int j8 = (idx & 31) << 3;    // half offset within row
            *(uint4*)&Th[i * TSTR + j8] = src[idx];
        }
    }

    const int ri = tid >> 4;          // row-in-block 0..15
    const int li = tid & 15;          // 16 threads per row
    const int r  = blockIdx.x * 16 + ri;   // global row = a*64 + b
    const int b  = r & 63;
    const int ai = r >> 6;
    const float ab = abil[ai];

    // alpha0 = init (alpha stored [row][state], 16 contiguous floats/thread)
#pragma unroll
    for (int u = 0; u < 16; ++u) {
        int s = li * 16 + u;
        Aal[ri * ASTR + s] = initv[s];
    }
    __syncthreads();

    const int lane = tid & 31;
    const int wv   = tid >> 5;
    const int n16  = lane & 15;
    const int hi   = lane >> 4;

    int p = pseq[b * NT + 0];
    int c = corr[b * NT + 0];

    for (int t = 0; t < NT; ++t) {
        // ---------------- Phase A: observation terms + row LSE ----------------
        // prefetch next timestep's problem-logit row into cache
        if (t + 1 < NT) {
            int pn = pseq[b * NT + t + 1];
            __builtin_prefetch(&PL[pn * NS + li * 16], 0, 0);
        }
        float x0[16], x1[16];
        float mx0 = -3.4e38f, mx1 = -3.4e38f;
        // 16 contiguous PL floats and 16 contiguous alpha floats per thread
        const float4* plv = (const float4*)(PL + p * NS + li * 16);
        const float4* alv = (const float4*)(Aal + ri * ASTR + li * 16);
#pragma unroll
        for (int q = 0; q < 4; ++q) {
            float4 plq = plv[q];
            float4 alq = alv[q];
            float pls[4] = {plq.x, plq.y, plq.z, plq.w};
            float als[4] = {alq.x, alq.y, alq.z, alq.w};
#pragma unroll
            for (int e = 0; e < 4; ++e) {
                int u = q * 4 + e;
                int s = li * 16 + u;
                float ocl = pls[e];
                if (s == 0) ocl += ab;
                else if (s == 1) ocl -= ab;
                float ls1 = logsigf(2.0f * ocl);      // obs_lp[...,1]
                float ls0 = ls1 - 2.0f * ocl;         // obs_lp[...,0]
                x0[u] = als[e] + ls0;
                x1[u] = als[e] + ls1;
                mx0 = fmaxf(mx0, x0[u]);
                mx1 = fmaxf(mx1, x1[u]);
            }
        }
        // cross-lane max within the 16-lane row group (wave32, xor<=8 stays in-group)
        for (int o = 8; o >= 1; o >>= 1) {
            mx0 = fmaxf(mx0, __shfl_xor(mx0, o, 32));
            mx1 = fmaxf(mx1, __shfl_xor(mx1, o, 32));
        }
        // Sum pass; the chosen-branch exponential IS the V-panel value:
        // v[s] = exp(x_c[s] - mx_c) -> store to LDS directly (no 3rd exp pass)
        float s0 = 0.0f, s1 = 0.0f;
#pragma unroll
        for (int u = 0; u < 16; ++u) {
            float e0 = expf(x0[u] - mx0);
            float e1 = expf(x1[u] - mx1);
            s0 += e0;
            s1 += e1;
            Vh[ri * VSTR + li * 16 + u] = (_Float16)(c ? e1 : e0);
        }
        for (int o = 8; o >= 1; o >>= 1) {
            s0 += __shfl_xor(s0, o, 32);
            s1 += __shfl_xor(s1, o, 32);
        }
        float lse0 = mx0 + logf(s0);
        float lse1 = mx1 + logf(s1);
        float mm = fmaxf(lse0, lse1);
        float logZ = mm + logf(expf(lse0 - mm) + expf(lse1 - mm));
        if (li == 0) {
            lpws[r * (NT * 2) + t * 2 + 0] = lse0 - logZ;
            lpws[r * (NT * 2) + t * 2 + 1] = lse1 - logZ;
            rowm[ri] = c ? mx1 : mx0;
        }
        // fetch next step's indices before the barrier (hides scalar latency)
        if (t + 1 < NT) {
            p = pseq[b * NT + t + 1];
            c = corr[b * NT + t + 1];
        }
        __syncthreads();

        // ---------------- Phase B: WMMA  alpha' = m + log(T @ v) ----------------
        float nal[16];
#pragma unroll
        for (int sub = 0; sub < 2; ++sub) {
            const int mt = wv * 2 + sub;       // state tile 0..15
            const int M = mt * 16 + n16;       // A-matrix row (new state)
            v8f acc = {0.f, 0.f, 0.f, 0.f, 0.f, 0.f, 0.f, 0.f};
#pragma unroll
            for (int kk = 0; kk < 8; ++kk) {
                const int kb = kk * 32;
                v16h af, bf;
                // A frag (16x32 f16): elems 0-7 = Th[M][kb+hi*8 ..+7],
                //                     elems 8-15 = Th[M][kb+16+hi*8 ..+7]
#pragma unroll
                for (int e = 0; e < 8; ++e) {
                    af[e]     = Th[M * TSTR + kb + hi * 8 + e];
                    af[8 + e] = Th[M * TSTR + kb + 16 + hi * 8 + e];
                }
                // B frag (32x16 f16): N = lane%16, K = kb + (e/8)*16 + hi*8 + e%8
                // V row-major -> two 8-half (16B) runs per lane
#pragma unroll
                for (int e = 0; e < 16; ++e) {
                    bf[e] = Vh[n16 * VSTR + kb + (e >> 3) * 16 + hi * 8 + (e & 7)];
                }
                acc = __builtin_amdgcn_wmma_f32_16x16x32_f16(
                    false, af, false, bf, (short)0, acc, false, false);
            }
            float rm = rowm[n16];
#pragma unroll
            for (int vp = 0; vp < 8; ++vp) {
                nal[sub * 8 + vp] = rm + logf(fmaxf(acc[vp], 1e-37f));
            }
        }
        // write alpha' back: C layout lane holds (M = vp + 8*hi, N = lane%16);
        // alpha is [row][state] -> 8 contiguous floats (32B) per sub-tile
#pragma unroll
        for (int sub = 0; sub < 2; ++sub) {
            int ibase = (wv * 2 + sub) * 16 + hi * 8;
#pragma unroll
            for (int vp = 0; vp < 8; ++vp) {
                Aal[n16 * ASTR + ibase + vp] = nal[sub * 8 + vp];
            }
        }
        __syncthreads();
    }
}

// ---------------------------------------------------------------------------
// Epilogue: posterior mixture over ability levels (tiny: 64 threads)
// ---------------------------------------------------------------------------
__global__ void k_post(const float* __restrict__ lpws,
                       const int* __restrict__ corr,
                       float* __restrict__ out) {
    int b = threadIdx.x;  // 0..63
    float cum[NA];
#pragma unroll
    for (int a = 0; a < NA; ++a) cum[a] = 0.0f;
    for (int t = 0; t < NT; ++t) {
        float mp = cum[0];
#pragma unroll
        for (int a = 1; a < NA; ++a) mp = fmaxf(mp, cum[a]);
        float Z = 0.0f;
#pragma unroll
        for (int a = 0; a < NA; ++a) Z += expf(cum[a] - mp);
        float logZ = mp + logf(Z);

        float lp0[NA], lp1[NA];
#pragma unroll
        for (int a = 0; a < NA; ++a) {
            const float* q = lpws + (a * NB + b) * (NT * 2) + t * 2;
            lp0[a] = q[0];
            lp1[a] = q[1];
        }
        float m0 = -3.4e38f, m1 = -3.4e38f;
#pragma unroll
        for (int a = 0; a < NA; ++a) {
            m0 = fmaxf(m0, cum[a] - logZ + lp0[a]);
            m1 = fmaxf(m1, cum[a] - logZ + lp1[a]);
        }
        float t0 = 0.0f, t1 = 0.0f;
#pragma unroll
        for (int a = 0; a < NA; ++a) {
            t0 += expf(cum[a] - logZ + lp0[a] - m0);
            t1 += expf(cum[a] - logZ + lp1[a] - m1);
        }
        out[b * (NT * 2) + t * 2 + 0] = m0 + logf(t0);
        out[b * (NT * 2) + t * 2 + 1] = m1 + logf(t1);

        int c = corr[b * NT + t];
#pragma unroll
        for (int a = 0; a < NA; ++a) cum[a] += c ? lp1[a] : lp0[a];
    }
}

extern "C" void kernel_launch(void* const* d_in, const int* in_sizes, int n_in,
                              void* d_out, int out_size, void* d_ws, size_t ws_size,
                              hipStream_t stream) {
    (void)in_sizes; (void)n_in; (void)out_size; (void)ws_size;
    const int*   corr  = (const int*)d_in[0];
    const int*   pseq  = (const int*)d_in[1];
    const int*   test  = (const int*)d_in[2];
    const float* off   = (const float*)d_in[3];
    const float* slope = (const float*)d_in[4];
    const float* ml    = (const float*)d_in[5];
    const float* pL    = (const float*)d_in[6];
    const float* pF    = (const float*)d_in[7];
    const float* pi    = (const float*)d_in[8];
    const float* abil  = (const float*)d_in[9];
    // d_in[10] decoder, d_in[11] tim: reconstructed via bit math on device
    float* out = (float*)d_out;

    // Workspace layout
    char* ws = (char*)d_ws;
    float*    PL_w   = (float*)ws;                               // 1000*256*4 = 1,024,000
    _Float16* Texp_w = (_Float16*)(ws + 1024000);                // 256*256*2  =   131,072
    float*    init_w = (float*)(ws + 1024000 + 131072);          // 256*4      =     1,024
    float*    lp_w   = (float*)(ws + 1024000 + 131072 + 1024);   // 320*200*2*4=   512,000

    k_problem_logits<<<NP, NS, 0, stream>>>(ml, off, slope, test, PL_w);
    k_trans<<<NS, NS, 0, stream>>>(pL, pF, Texp_w);
    k_init<<<1, NS, 0, stream>>>(pi, init_w);

    size_t smem = (size_t)NS * TSTR * 2   // T matrix f16 (132 KB)
                + (size_t)16 * VSTR * 2   // V panel f16
                + (size_t)16 * ASTR * 4   // alpha f32
                + 16 * 4;                 // row maxes
    k_forward<<<NROWS / 16, 256, smem, stream>>>(corr, pseq, abil, PL_w, Texp_w,
                                                 init_w, lp_w);

    k_post<<<1, NB, 0, stream>>>(lp_w, corr, out);
}
